// DecoderLayer_11424613007924
// MI455X (gfx1250) — matrix operand order repaired and checked by reading, not compiled, not run on hardware
//
#include <hip/hip_runtime.h>

#define NEGB (-1000000000.0f)

typedef __attribute__((ext_vector_type(16))) __bf16 v16bf;
typedef __attribute__((ext_vector_type(8)))  __bf16 v8bf;
typedef __attribute__((ext_vector_type(8)))  float  v8f;
typedef __attribute__((ext_vector_type(4)))  unsigned v4u;
typedef __attribute__((ext_vector_type(8)))  int    v8i;
typedef __attribute__((ext_vector_type(4)))  int    v4i;

union BF16x16 { v16bf v; v8bf h[2]; };
union BF8U    { v8bf v; __bf16 e[8]; };

#if __has_builtin(__builtin_amdgcn_tensor_load_to_lds) && \
    __has_builtin(__builtin_amdgcn_s_wait_tensorcnt)
#define HAS_TDM 1
#else
#define HAS_TDM 0
#endif

static __device__ __forceinline__ __bf16 f2bf(float f) {
  unsigned u = __builtin_bit_cast(unsigned, f);
  unsigned r = (u + 0x7FFFu + ((u >> 16) & 1u)) >> 16;
  unsigned short s = (unsigned short)r;
  return __builtin_bit_cast(__bf16, s);
}

static __device__ __forceinline__ int iabs_(int x) { return x < 0 ? -x : x; }

static __device__ __forceinline__ v8f wmma_bf16(v16bf a, v16bf b, v8f c) {
  return __builtin_amdgcn_wmma_f32_16x16x32_bf16(false, a, false, b, (short)0, c,
                                                 false, false);
}

static __device__ __forceinline__ float rmax16(float v) {
  v = fmaxf(v, __shfl_xor(v, 1, 32));
  v = fmaxf(v, __shfl_xor(v, 2, 32));
  v = fmaxf(v, __shfl_xor(v, 4, 32));
  v = fmaxf(v, __shfl_xor(v, 8, 32));
  return v;
}
static __device__ __forceinline__ float rsum16(float v) {
  v += __shfl_xor(v, 1, 32);
  v += __shfl_xor(v, 2, 32);
  v += __shfl_xor(v, 4, 32);
  v += __shfl_xor(v, 8, 32);
  return v;
}
static __device__ __forceinline__ float rsum32(float v) {
  v += __shfl_xor(v, 1, 32);
  v += __shfl_xor(v, 2, 32);
  v += __shfl_xor(v, 4, 32);
  v += __shfl_xor(v, 8, 32);
  v += __shfl_xor(v, 16, 32);
  return v;
}

// ---------------------------------------------------------------------------
// TDM: issue one TENSOR_LOAD_TO_LDS for a 2D bf16 tile (rows x dim0 elems,
// global row stride in elements). rows==1 => 1D copy. D# per ISA 08 §8.3/8.4:
//   group0: count=1 | lds_addr | global_addr(57b) | type=2
//   group1: data_size=2B | tensor_dim0/1 | tile_dim0/1 | dim0_stride
// Wave-level op (EXEC ignored) -> call from one wave under a scalar branch.
// ---------------------------------------------------------------------------
static __device__ __forceinline__ void tdm_load_tile_bf16(
    const void* gsrc, void* lds_dst, unsigned dim0, unsigned rows,
    unsigned long long row_stride_elems)
{
#if HAS_TDM
  const unsigned long long ga = (unsigned long long)gsrc;
  const unsigned lds_off = (unsigned)(unsigned long long)lds_dst;
  v4u g0;
  g0[0] = 1u;                                            // count=1, user mode
  g0[1] = lds_off;                                       // LDS byte address
  g0[2] = (unsigned)(ga & 0xFFFFFFFFu);                  // global_addr[31:0]
  g0[3] = (unsigned)((ga >> 32) & 0x1FFFFFFu) | 0x80000000u;  // addr[56:32]|type=2
  v8i g1;
  g1[0] = 0x00010000;                                    // data_size=1 (2 bytes)
  g1[1] = (int)((dim0 & 0xFFFFu) << 16);                 // tensor_dim0[15:0]
  g1[2] = (int)((dim0 >> 16) | ((rows & 0xFFFFu) << 16)); // dim0 hi | tensor_dim1 lo
  g1[3] = (int)((dim0 & 0xFFFFu) << 16);                 // dim1 hi=0 | tile_dim0
  g1[4] = (int)((rows > 1u) ? (rows & 0xFFFFu) : 0u);    // tile_dim1 (0 => 1D)
  g1[5] = (int)(unsigned)(row_stride_elems & 0xFFFFFFFFull);      // stride[31:0]
  g1[6] = (int)(unsigned)((row_stride_elems >> 32) & 0xFFFFull);  // stride[47:32]
  g1[7] = 0;
  v4i z4; z4[0] = 0; z4[1] = 0; z4[2] = 0; z4[3] = 0;
#if __clang_major__ >= 23
  v8i z8;
#pragma unroll
  for (int i = 0; i < 8; ++i) z8[i] = 0;
  __builtin_amdgcn_tensor_load_to_lds(g0, g1, z4, z4, z8, 0);
#else
  __builtin_amdgcn_tensor_load_to_lds(g0, g1, z4, z4, 0);
#endif
#else
  (void)gsrc; (void)lds_dst; (void)dim0; (void)rows; (void)row_stride_elems;
#endif
}

static __device__ __forceinline__ void tdm_wait0() {
#if HAS_TDM
  __builtin_amdgcn_s_wait_tensorcnt(0);
#endif
}

// ---------------------------------------------------------------------------
// GEMM inner loop: C(64x64 block) = A(f32 MxK row-major) * B^T(bf16 NxK).
// 8 waves, wave tile 16x32 => two 16x16x32 bf16 WMMAs per 32-wide K step.
// A tile converted f32->bf16 by all threads; B tile DMA'd by the TDM
// (contiguous NxK sub-block, 64 rows x 32 elems, stride K).
// ---------------------------------------------------------------------------
static __device__ __forceinline__ void gemm_core(
    const float* __restrict__ A, const __bf16* __restrict__ BT,
    int K, int m0, int n0, __bf16* As, __bf16* Bs, v8f& acc0, v8f& acc1)
{
  const int t = threadIdx.x;
  const int lane = t & 31, wave = t >> 5;
  const int lhalf = lane >> 4, lmod = lane & 15;
  const int mw = wave & 3, nw = wave >> 2;
  const int a_r = t >> 2, a_kc = t & 3;   // A loader: row 0..63, 8-float chunk

#pragma unroll 1
  for (int k0 = 0; k0 < K; k0 += 32) {
    __syncthreads();
    {
      const float* ap = A + (long)(m0 + a_r) * K + k0 + a_kc * 8;
      float4 f0 = *(const float4*)ap;
      float4 f1 = *(const float4*)(ap + 4);
      BF8U av;
      av.e[0] = f2bf(f0.x); av.e[1] = f2bf(f0.y);
      av.e[2] = f2bf(f0.z); av.e[3] = f2bf(f0.w);
      av.e[4] = f2bf(f1.x); av.e[5] = f2bf(f1.y);
      av.e[6] = f2bf(f1.z); av.e[7] = f2bf(f1.w);
      *(v8bf*)&As[a_r * 40 + a_kc * 8] = av.v;
      if (k0 + 32 < K) __builtin_prefetch(ap + 32, 0, 0);
    }
#if HAS_TDM
    if (wave == 0) {
      tdm_load_tile_bf16(BT + (long)n0 * K + k0, Bs, 32u, 64u,
                         (unsigned long long)K);
      tdm_wait0();
    }
#else
    {
      const int b_n = t >> 2, b_c = t & 3;  // 64 rows x 4 chunks of 8
      *(v8bf*)&Bs[b_n * 32 + b_c * 8] =
          *(const v8bf*)&BT[(long)(n0 + b_n) * K + k0 + b_c * 8];
    }
#endif
    __syncthreads();

    BF16x16 a, b0, b1;
    const __bf16* arp = &As[(mw * 16 + lmod) * 40 + lhalf * 16];
    a.h[0] = *(const v8bf*)arp;
    a.h[1] = *(const v8bf*)(arp + 8);
    const __bf16* b0p = &Bs[(nw * 32 + lmod) * 32 + lhalf * 16];
    b0.h[0] = *(const v8bf*)b0p;
    b0.h[1] = *(const v8bf*)(b0p + 8);
    const __bf16* b1p = &Bs[(nw * 32 + 16 + lmod) * 32 + lhalf * 16];
    b1.h[0] = *(const v8bf*)b1p;
    b1.h[1] = *(const v8bf*)(b1p + 8);

    acc0 = wmma_bf16(a.v, b0.v, acc0);
    acc1 = wmma_bf16(a.v, b1.v, acc1);
  }
}

// Generic GEMM with fused bias / residual / relu epilogue. C is f32 MxN.
__global__ __launch_bounds__(256) void gemm_kernel(
    const float* __restrict__ A, const __bf16* __restrict__ BT,
    const float* __restrict__ bias, const float* __restrict__ resid,
    float* __restrict__ C, int N, int K, int relu)
{
  __shared__ alignas(16) __bf16 As[64 * 40];
  __shared__ alignas(16) __bf16 Bs[64 * 32];
  const int m0 = blockIdx.y * 64, n0 = blockIdx.x * 64;
  v8f acc0, acc1;
#pragma unroll
  for (int r = 0; r < 8; ++r) { acc0[r] = 0.f; acc1[r] = 0.f; }
  gemm_core(A, BT, K, m0, n0, As, Bs, acc0, acc1);

  const int lane = threadIdx.x & 31, wave = threadIdx.x >> 5;
  const int lhalf = lane >> 4, lmod = lane & 15;
  const int mw = wave & 3, nw = wave >> 2;
#pragma unroll
  for (int r = 0; r < 8; ++r) {
    const int m = m0 + mw * 16 + r + lhalf * 8;
#pragma unroll
    for (int j = 0; j < 2; ++j) {
      const int n = n0 + nw * 32 + j * 16 + lmod;
      float v = (j == 0) ? acc0[r] : acc1[r];
      if (bias)  v += bias[n];
      if (resid) v += resid[(long)m * N + n];
      if (relu)  v = fmaxf(v, 0.f);
      C[(long)m * N + n] = v;
    }
  }
}

// QKV projection: custom epilogue scattering bf16 outputs:
//   Q (b,h,l,dh) scaled by dh^-0.5, K (b,h,l,dh), V transposed (b,h,dh,l).
__global__ __launch_bounds__(256) void gemm_qkv_kernel(
    const float* __restrict__ A, const __bf16* __restrict__ BT,
    __bf16* __restrict__ Qb, __bf16* __restrict__ Kb, __bf16* __restrict__ Vb)
{
  __shared__ alignas(16) __bf16 As[64 * 40];
  __shared__ alignas(16) __bf16 Bs[64 * 32];
  const int m0 = blockIdx.y * 64, n0 = blockIdx.x * 64;
  v8f acc0, acc1;
#pragma unroll
  for (int r = 0; r < 8; ++r) { acc0[r] = 0.f; acc1[r] = 0.f; }
  gemm_core(A, BT, 256, m0, n0, As, Bs, acc0, acc1);

  const int lane = threadIdx.x & 31, wave = threadIdx.x >> 5;
  const int lhalf = lane >> 4, lmod = lane & 15;
  const int mw = wave & 3, nw = wave >> 2;
  const float qscale = 0.17677669529663687f;  // 1/sqrt(32)
#pragma unroll
  for (int r = 0; r < 8; ++r) {
    const int m = m0 + mw * 16 + r + lhalf * 8;
    const int b = m >> 10, l = m & 1023;
#pragma unroll
    for (int j = 0; j < 2; ++j) {
      const int n = n0 + nw * 32 + j * 16 + lmod;
      const int which = n >> 8;
      const int d = n & 255, hh = d >> 5, dd = d & 31;
      const long bh = (long)(b * 8 + hh);
      float v = (j == 0) ? acc0[r] : acc1[r];
      if (which == 0) {
        Qb[(bh * 1024 + l) * 32 + dd] = f2bf(v * qscale);
      } else if (which == 1) {
        Kb[(bh * 1024 + l) * 32 + dd] = f2bf(v);
      } else {
        Vb[(bh * 32 + dd) * 1024 + l] = f2bf(v);  // transposed for P@V B-frags
      }
    }
  }
}

// ---------------------------------------------------------------------------
// Block-local attention. grid = B*H*4 blocks of 256 threads (8 waves).
// K tile: 1D TDM copy (row-major bf16). V tile: 2D TDM copy of the
// pre-transposed (dh, l) layout. Streaming softmax over 32-key chunks;
// 4 bf16 WMMAs per chunk per wave tile.
// ---------------------------------------------------------------------------
__global__ __launch_bounds__(256) void attn_kernel(
    const __bf16* __restrict__ Qg, const __bf16* __restrict__ Kg,
    const __bf16* __restrict__ Vg, float* __restrict__ Ag)
{
  extern __shared__ char smem[];
  __bf16* Ks = (__bf16*)smem;                   // Lk x 32, stride 32
  __bf16* Vt = (__bf16*)(smem + 32768);         // 32 x Lk, stride Lk
  __bf16* Ps = (__bf16*)(smem + 65536);         // per-wave 16x32, stride 40

  const int blk  = blockIdx.x;
  const int qblk = blk & 3;
  const int h    = (blk >> 2) & 7;
  const int b    = blk >> 5;
  const int t    = threadIdx.x;
  const int wave = t >> 5, lane = t & 31;
  const int lhalf = lane >> 4, lmod = lane & 15;
  const bool first = (qblk == 0);
  const int Lk     = first ? 256 : 512;
  const int kstart = first ? 0 : (qblk - 1) * 256;
  const int qstart = qblk * 256;
  const long base_bh = ((long)(b * 8 + h)) * 1024;

  const __bf16* Kp = Kg + (base_bh + kstart) * 32;          // (l, dh) rows
  const __bf16* Vp = Vg + (long)(b * 8 + h) * 32 * 1024 + kstart; // (dh, l)

#if HAS_TDM
  if (wave == 0) {
    tdm_load_tile_bf16(Kp, Ks, (unsigned)(Lk * 32), 1u, 0ull);
    tdm_load_tile_bf16(Vp, Vt, (unsigned)Lk, 32u, 1024ull);
    tdm_wait0();
  }
#else
  for (int i = t; i < (Lk * 32) / 8; i += 256)
    *(v8bf*)&Ks[i * 8] = *(const v8bf*)&Kp[i * 8];
  {
    const int cpr = Lk / 8;  // 8-elem chunks per row
    for (int i = t; i < 32 * cpr; i += 256) {
      const int row = i / cpr, c = (i % cpr) * 8;
      *(v8bf*)&Vt[row * Lk + c] = *(const v8bf*)&Vp[(long)row * 1024 + c];
    }
  }
#endif
  __syncthreads();

  const __bf16* Qrow0 = Qg + (base_bh + qstart) * 32;
  __bf16* Pw = Ps + wave * 16 * 40;

#pragma unroll 1
  for (int mt = 0; mt < 2; ++mt) {
    const int qtb = wave * 32 + mt * 16;  // query tile base (0..255)

    // Q A-fragment straight from global bf16.
    BF16x16 aq;
    {
      const __bf16* qr = Qrow0 + (long)(qtb + lmod) * 32 + lhalf * 16;
      aq.h[0] = *(const v8bf*)qr;
      aq.h[1] = *(const v8bf*)(qr + 8);
    }

    v8f acc0, acc1;
    float mrow[8], srow[8];
#pragma unroll
    for (int r = 0; r < 8; ++r) {
      acc0[r] = 0.f; acc1[r] = 0.f; mrow[r] = -1e30f; srow[r] = 0.f;
    }

#pragma unroll 1
    for (int kc = 0; kc < Lk; kc += 32) {
      BF16x16 bk0, bk1;
      {
        const __bf16* k0p = &Ks[(kc + lmod) * 32 + lhalf * 16];
        bk0.h[0] = *(const v8bf*)k0p;  bk0.h[1] = *(const v8bf*)(k0p + 8);
        const __bf16* k1p = &Ks[(kc + 16 + lmod) * 32 + lhalf * 16];
        bk1.h[0] = *(const v8bf*)k1p;  bk1.h[1] = *(const v8bf*)(k1p + 8);
      }
      v8f z;
#pragma unroll
      for (int r = 0; r < 8; ++r) z[r] = 0.f;
      v8f s0 = wmma_bf16(aq.v, bk0.v, z);
      v8f s1 = wmma_bf16(aq.v, bk1.v, z);

      const int n0k = kc + lmod, n1k = kc + 16 + lmod;
#pragma unroll
      for (int r = 0; r < 8; ++r) {
        const int q = qtb + r + lhalf * 8;
        float b0, b1;
        if (first) {
          b0 = (n0k <= q) ? 0.f : NEGB;
          b1 = (n1k <= q) ? 0.f : NEGB;
        } else {
          const int cp = 256 + q, ch = cp >> 5, cw = cp & 31;
          int oh = n0k >> 5, ow = n0k & 31;
          b0 = ((n0k <= cp) && (iabs_(ch - oh) <= 6) && (iabs_(cw - ow) <= 6))
                   ? 0.f : NEGB;
          oh = n1k >> 5; ow = n1k & 31;
          b1 = ((n1k <= cp) && (iabs_(ch - oh) <= 6) && (iabs_(cw - ow) <= 6))
                   ? 0.f : NEGB;
        }
        const float v0 = s0[r] + b0;
        const float v1 = s1[r] + b1;
        const float cmax = rmax16(fmaxf(v0, v1));
        const float nm   = fmaxf(mrow[r], cmax);
        const float sc   = __expf(mrow[r] - nm);
        const float p0   = __expf(v0 - nm);
        const float p1   = __expf(v1 - nm);
        srow[r] = srow[r] * sc + rsum16(p0 + p1);
        acc0[r] *= sc;
        acc1[r] *= sc;
        mrow[r] = nm;
        const int prow = r + lhalf * 8;
        Pw[prow * 40 + lmod]      = f2bf(p0);
        Pw[prow * 40 + 16 + lmod] = f2bf(p1);
      }

      // Reshape P (C layout) -> A fragment via per-wave LDS staging.
      BF16x16 ap;
      {
        const __bf16* pp = &Pw[lmod * 40 + lhalf * 16];
        ap.h[0] = *(const v8bf*)pp;
        ap.h[1] = *(const v8bf*)(pp + 8);
      }
      // V B-fragments from (dh, l) LDS: col = dh, 16 contiguous keys/half.
      BF16x16 bv0, bv1;
      {
        const __bf16* v0p = &Vt[lmod * Lk + kc + lhalf * 16];
        bv0.h[0] = *(const v8bf*)v0p;  bv0.h[1] = *(const v8bf*)(v0p + 8);
        const __bf16* v1p = &Vt[(16 + lmod) * Lk + kc + lhalf * 16];
        bv1.h[0] = *(const v8bf*)v1p;  bv1.h[1] = *(const v8bf*)(v1p + 8);
      }
      acc0 = wmma_bf16(ap.v, bv0.v, acc0);
      acc1 = wmma_bf16(ap.v, bv1.v, acc1);
    }

    // Normalize and write f32 attn output in (b, l, h*32+dd) layout.
#pragma unroll
    for (int r = 0; r < 8; ++r) {
      const float inv = 1.f / srow[r];
      const int q = qtb + r + lhalf * 8;
      float* dst = Ag + ((long)b * 1024 + qstart + q) * 256 + h * 32;
      dst[lmod]      = acc0[r] * inv;
      dst[16 + lmod] = acc1[r] * inv;
    }
  }
}

// LayerNorm over rows of 256. One wave per row, 8 rows per block.
__global__ __launch_bounds__(256) void ln_kernel(
    const float* __restrict__ X, const float* __restrict__ g,
    const float* __restrict__ bb, float* __restrict__ out)
{
  const int row  = blockIdx.x * 8 + (threadIdx.x >> 5);
  const int lane = threadIdx.x & 31;
  const float* x = X + (long)row * 256 + lane * 8;
  float4 u0 = *(const float4*)x;
  float4 u1 = *(const float4*)(x + 4);
  float v[8] = {u0.x, u0.y, u0.z, u0.w, u1.x, u1.y, u1.z, u1.w};
  float s = 0.f;
#pragma unroll
  for (int e = 0; e < 8; ++e) s += v[e];
  const float mean = rsum32(s) * (1.f / 256.f);
  float q = 0.f;
#pragma unroll
  for (int e = 0; e < 8; ++e) { const float d = v[e] - mean; q += d * d; }
  const float rstd = rsqrtf(rsum32(q) * (1.f / 256.f) + 1e-6f);
  float* o = out + (long)row * 256 + lane * 8;
#pragma unroll
  for (int e = 0; e < 8; ++e)
    o[e] = (v[e] - mean) * rstd * g[lane * 8 + e] + bb[lane * 8 + e];
}

// Transposed weight conversion: dst (NxK bf16) = src(KxN f32)^T.
__global__ void cvt_wT_kernel(const float* __restrict__ src,
                              __bf16* __restrict__ dst, int N, int K)
{
  const int i = blockIdx.x * 256 + threadIdx.x;
  if (i < N * K) {
    const int n = i / K, k = i % K;
    dst[i] = f2bf(src[(long)k * N + n]);
  }
}

// Pack (Wq|Wk|Wv)^T into a single 768x256 bf16 matrix.
__global__ void cvt_qkvT_kernel(const float* __restrict__ Wq,
                                const float* __restrict__ Wk,
                                const float* __restrict__ Wv,
                                __bf16* __restrict__ dst)
{
  const int i = blockIdx.x * 256 + threadIdx.x;  // < 768*256
  const int n = i >> 8, k = i & 255;
  const int which = n >> 8, d = n & 255;
  const float* s = (which == 0) ? Wq : ((which == 1) ? Wk : Wv);
  dst[i] = f2bf(s[k * 256 + d]);
}

extern "C" void kernel_launch(void* const* d_in, const int* in_sizes, int n_in,
                              void* d_out, int out_size, void* d_ws, size_t ws_size,
                              hipStream_t stream) {
  const float* X     = (const float*)d_in[0];
  const float* Wq    = (const float*)d_in[1];
  const float* Wk    = (const float*)d_in[2];
  const float* Wv    = (const float*)d_in[3];
  const float* Wo    = (const float*)d_in[4];
  const float* ln1_g = (const float*)d_in[5];
  const float* ln1_b = (const float*)d_in[6];
  const float* W1    = (const float*)d_in[7];
  const float* b1    = (const float*)d_in[8];
  const float* W2    = (const float*)d_in[9];
  const float* b2    = (const float*)d_in[10];
  const float* ln2_g = (const float*)d_in[11];
  const float* ln2_b = (const float*)d_in[12];

  const int M  = in_sizes[0] / 256;   // B*L = 32768
  const int Bn = M / 1024;            // batches = 32

  char* ws = (char*)d_ws;
  __bf16* WqkvT = (__bf16*)ws;                              // 768x256 bf16
  __bf16* WoT   = (__bf16*)(ws + 393216);                   // 256x256
  __bf16* W1T   = (__bf16*)(ws + 393216 + 131072);          // 512x256
  __bf16* W2T   = (__bf16*)(ws + 393216 + 131072 + 262144); // 256x512

  const size_t off = 1u << 20;                   // 1 MB weight region
  const size_t HB  = (size_t)M * 256 * 2;        // bf16 activation plane
  const size_t FB  = (size_t)M * 256 * 4;        // f32 activation plane
  __bf16* Qb = (__bf16*)(ws + off);              // (b,h,l,dh) bf16
  __bf16* Kb = (__bf16*)(ws + off + HB);
  __bf16* Vb = (__bf16*)(ws + off + 2 * HB);     // (b,h,dh,l) bf16
  float*  Aa = (float*)(ws + off + 3 * HB);      // attn out (M,256) f32
  float*  R1 = (float*)(ws + off + 3 * HB + FB); // pre-LN1; reused pre-LN2
  float*  X1 = (float*)(ws + off + 3 * HB + 2 * FB);
  float*  Hf = (float*)(ws + off);               // FFN hidden (M,512) f32,
                                                 // aliases dead Q/K/V/Aa

  cvt_qkvT_kernel<<<768, 256, 0, stream>>>(Wq, Wk, Wv, WqkvT);
  cvt_wT_kernel<<<256, 256, 0, stream>>>(Wo, WoT, 256, 256);
  cvt_wT_kernel<<<512, 256, 0, stream>>>(W1, W1T, 512, 256);
  cvt_wT_kernel<<<512, 256, 0, stream>>>(W2, W2T, 256, 512);

  // QKV projection (Q pre-scaled by dh^-0.5; V stored transposed).
  gemm_qkv_kernel<<<dim3(768 / 64, M / 64), 256, 0, stream>>>(X, WqkvT, Qb, Kb, Vb);

  // Block-local attention (TDM-staged K/V tiles).
  const int attn_smem = 32768 + 32768 + 10240;  // Ks + Vt + P staging
  attn_kernel<<<Bn * 8 * 4, 256, attn_smem, stream>>>(Qb, Kb, Vb, Aa);

  // O projection + residual X -> R1, then LN1 -> X1.
  gemm_kernel<<<dim3(256 / 64, M / 64), 256, 0, stream>>>(
      Aa, WoT, nullptr, X, R1, 256, 256, 0);
  ln_kernel<<<M / 8, 256, 0, stream>>>(R1, ln1_g, ln1_b, X1);

  // FFN.
  gemm_kernel<<<dim3(512 / 64, M / 64), 256, 0, stream>>>(
      X1, W1T, b1, nullptr, Hf, 512, 256, 1);
  gemm_kernel<<<dim3(256 / 64, M / 64), 256, 0, stream>>>(
      Hf, W2T, b2, X1, R1, 256, 512, 0);
  ln_kernel<<<M / 8, 256, 0, stream>>>(R1, ln2_g, ln2_b, (float*)d_out);
}
